// MultiQueryAttention_39324720562239
// MI455X (gfx1250) — matrix operand order repaired and checked by reading, not compiled, not run on hardware
//
#include <hip/hip_runtime.h>
#include <cstdint>
#include <cstddef>

// ---------------------------------------------------------------------------
// MQA forward for gfx1250 (MI455X): bf16 WMMA + async global->LDS staging.
//   B=2, S=2048, HID=2048, H=16, D=128  (single KV head)
// ---------------------------------------------------------------------------

typedef __attribute__((ext_vector_type(16))) __bf16 v16bf;
typedef __attribute__((ext_vector_type(8)))  float  v8f;
typedef __attribute__((ext_vector_type(4)))  unsigned int u32x4;

union Frag {
    v16bf v;
    u32x4 q[2];
};

__device__ __forceinline__ unsigned short to_bf16(float f) {
    unsigned u = __float_as_uint(f);
    unsigned r = (u + 0x7FFFu + ((u >> 16) & 1u)) >> 16;
    return (unsigned short)r;
}

__device__ __forceinline__ v8f wmma_bf16(v16bf a, v16bf b, v8f c) {
    // D = A(16x32 bf16) * B(32x16 bf16) + C(16x16 f32)
    return __builtin_amdgcn_wmma_f32_16x16x32_bf16(
        /*neg_a=*/false, a, /*neg_b=*/false, b,
        /*c_mod=*/(short)0, c, /*reuse_a=*/false, /*reuse_b=*/false);
}

// Low 32 bits of a flat shared-aperture address == LDS byte offset (ISA 10.2).
__device__ __forceinline__ unsigned lds_off(const void* p) {
    return (unsigned)(unsigned long long)p;
}

// Per-lane async copy: 16B global -> LDS, tracked by ASYNCcnt.
__device__ __forceinline__ void async_b128(unsigned lds_addr, const void* gaddr) {
    asm volatile("global_load_async_to_lds_b128 %0, %1, off"
                 :: "v"(lds_addr), "v"(gaddr)
                 : "memory");
}
__device__ __forceinline__ void wait_async0() {
    asm volatile("s_wait_asynccnt 0" ::: "memory");
}

// ---------------------------------------------------------------------------
// fp32 -> bf16 conversion
// ---------------------------------------------------------------------------
__global__ void __launch_bounds__(256) cvt_bf16_kernel(const float* __restrict__ src,
                                                       unsigned short* __restrict__ dst,
                                                       int n) {
    int i = (blockIdx.x * 256 + threadIdx.x) * 4;
    if (i + 3 < n) {
        float4 f = *(const float4*)(src + i);
        dst[i + 0] = to_bf16(f.x);
        dst[i + 1] = to_bf16(f.y);
        dst[i + 2] = to_bf16(f.z);
        dst[i + 3] = to_bf16(f.w);
    } else {
        for (int k = i; k < n; ++k) dst[k] = to_bf16(src[k]);
    }
}

// ---------------------------------------------------------------------------
// Generic GEMM:  C[M,N] = A[M,K](bf16) * W[N,K]^T (bf16) + bias
// 256 threads = 8 waves, tile 128(M) x 64(N), K-step 32.
// Double-buffered LDS, filled by global_load_async_to_lds_b128.
// ---------------------------------------------------------------------------
__global__ void __launch_bounds__(256) gemm_bf16_kernel(
    const unsigned short* __restrict__ A,   // [M,K] bf16 raw bits
    const unsigned short* __restrict__ W,   // [N,K] bf16 raw bits
    const float* __restrict__ bias,         // [N]
    unsigned short* __restrict__ Cb,        // bf16 out (or null)
    float* __restrict__ Cf,                 // f32 out (or null)
    int M, int N, int K) {
    __shared__ unsigned short At[2][128][32];
    __shared__ unsigned short Bt[2][64][32];

    const int tid  = threadIdx.x;
    const int wave = tid >> 5;
    const int lane = tid & 31;
    const int h    = lane >> 4;   // lane half (0/1)
    const int ln   = lane & 15;
    const int Mbase = blockIdx.x * 128;
    const int Nbase = blockIdx.y * 64;

    // staging addresses (per thread): A tile 128x32 (32B/thread), B tile 64x32 (16B/thread)
    const int rowA = tid >> 1, halfA = tid & 1;
    const unsigned short* gA = A + (size_t)(Mbase + rowA) * K + halfA * 16;
    const int nB = tid >> 2, koffB = (tid & 3) * 8;
    const unsigned short* gB = W + (size_t)(Nbase + nB) * K + koffB;
    unsigned ldsA[2], ldsB[2];
#pragma unroll
    for (int bb = 0; bb < 2; ++bb) {
        ldsA[bb] = lds_off(&At[bb][rowA][halfA * 16]);
        ldsB[bb] = lds_off(&Bt[bb][nB][koffB]);
    }

    auto stage = [&](int buf, int kk) {
        async_b128(ldsA[buf],      gA + kk);
        async_b128(ldsA[buf] + 16, gA + kk + 8);
        async_b128(ldsB[buf],      gB + kk);
    };

    v8f acc[4] = {};

    stage(0, 0);
    wait_async0();
    __syncthreads();

    const int nsteps = K >> 5;
    for (int s = 0; s < nsteps; ++s) {
        const int cur = s & 1;
        if (s + 1 < nsteps) stage(cur ^ 1, (s + 1) * 32);  // prefetch next tile

        // load all fragments first, then issue independent WMMAs back-to-back
        Frag a, bfr[4];
        const int arow = wave * 16 + ln;
        a.q[0] = *(const u32x4*)(&At[cur][arow][h * 8]);
        a.q[1] = *(const u32x4*)(&At[cur][arow][16 + h * 8]);
#pragma unroll
        for (int nt = 0; nt < 4; ++nt) {
            const unsigned short* br = &Bt[cur][nt * 16 + ln][h * 8];
            bfr[nt].q[0] = *(const u32x4*)(br);
            bfr[nt].q[1] = *(const u32x4*)(br + 16);
        }
#pragma unroll
        for (int nt = 0; nt < 4; ++nt)
            acc[nt] = wmma_bf16(a.v, bfr[nt].v, acc[nt]);

        wait_async0();
        __syncthreads();
    }

    // epilogue: C layout -> row = r + 8*h, col = ln
#pragma unroll
    for (int nt = 0; nt < 4; ++nt) {
        const int col = Nbase + nt * 16 + ln;
        const float bv = bias ? bias[col] : 0.0f;
#pragma unroll
        for (int r = 0; r < 8; ++r) {
            const int row = Mbase + wave * 16 + r + h * 8;
            const float val = acc[nt][r] + bv;
            if (Cf) Cf[(size_t)row * N + col] = val;
            if (Cb) Cb[(size_t)row * N + col] = to_bf16(val);
        }
    }
}

// ---------------------------------------------------------------------------
// Flash attention (causal, single KV head).
// Grid: (S/64, H, B). Block: 128 threads = 4 waves. Wave handles 16 q-rows.
// ---------------------------------------------------------------------------
__global__ void __launch_bounds__(128) mqa_attn_kernel(
    const unsigned short* __restrict__ Q,   // [B*S, 2048] bf16 (head-major cols)
    const unsigned short* __restrict__ Kb,  // [B*S, 128]  bf16
    const unsigned short* __restrict__ Vb,  // [B*S, 128]  bf16
    unsigned short* __restrict__ Ctx) {     // [B*S, 2048] bf16
    const int S = 2048, D = 128, HID = 2048;
    __shared__ unsigned short Kt[64][128];   // [key][d]
    __shared__ unsigned short Vt[128][64];   // [d][key]   (transposed)
    __shared__ unsigned short Pt[4][16][64]; // per-wave P scratch

    const int tid  = threadIdx.x;
    const int wave = tid >> 5;
    const int lane = tid & 31;
    const int h    = lane >> 4;
    const int ln   = lane & 15;
    const int qt   = blockIdx.x;      // query tile (64 rows)
    const int head = blockIdx.y;
    const int b    = blockIdx.z;
    const int qbase = qt * 64;

    // Q fragments (16 rows x 128 d) : 4 chunks of K=32, kept in registers
    Frag qf[4];
    {
        const unsigned short* qrow =
            Q + (size_t)(b * S + qbase + wave * 16 + ln) * HID + head * D;
#pragma unroll
        for (int c = 0; c < 4; ++c) {
            qf[c].q[0] = *(const u32x4*)(qrow + c * 32 + h * 8);
            qf[c].q[1] = *(const u32x4*)(qrow + c * 32 + 16 + h * 8);
        }
    }

    // per-thread K staging addresses (row = tid/2, 64 elems per half-row)
    const int krow = tid >> 1, koff = (tid & 1) * 64;
    const unsigned kldsa = lds_off(&Kt[krow][koff]);

    v8f o[8] = {};
    float mi[8], li[8];
#pragma unroll
    for (int r = 0; r < 8; ++r) { mi[r] = -1e30f; li[r] = 0.0f; }
    const float SC = 0.08838834764831845f;  // 1/sqrt(128)

    const int ntiles = qt + 1;  // causal: only tiles at/below diagonal
    for (int kt = 0; kt < ntiles; ++kt) {
        const int kbase = kt * 64;
        __syncthreads();  // previous tile's LDS reads complete
        {   // stage K tile [64][128] via async global->LDS (8x b128 per thread)
            const unsigned short* src = Kb + (size_t)(b * S + kbase + krow) * D + koff;
#pragma unroll
            for (int i = 0; i < 8; ++i) async_b128(kldsa + i * 16, src + i * 8);
        }
        {   // stage V transposed [d][key]
            int key = tid >> 1, dbase = (tid & 1) * 64;
            const unsigned short* src = Vb + (size_t)(b * S + kbase + key) * D + dbase;
#pragma unroll
            for (int i = 0; i < 64; ++i) Vt[dbase + i][key] = src[i];
        }
        wait_async0();
        __syncthreads();

        // scores S(16x64) = Q(16x128) * K^T  (c outer -> 4 independent WMMAs)
        v8f s[4] = {};
#pragma unroll
        for (int c = 0; c < 4; ++c) {
            Frag kf[4];
#pragma unroll
            for (int nt = 0; nt < 4; ++nt) {
                const unsigned short* kr = &Kt[nt * 16 + ln][c * 32 + h * 8];
                kf[nt].q[0] = *(const u32x4*)(kr);
                kf[nt].q[1] = *(const u32x4*)(kr + 16);
            }
#pragma unroll
            for (int nt = 0; nt < 4; ++nt)
                s[nt] = wmma_bf16(qf[c].v, kf[nt].v, s[nt]);
        }

        // scale + causal mask (diagonal tile only)
#pragma unroll
        for (int nt = 0; nt < 4; ++nt) {
#pragma unroll
            for (int r = 0; r < 8; ++r) {
                float x = s[nt][r] * SC;
                if (kt == qt) {
                    const int row = qbase + wave * 16 + r + h * 8;
                    const int col = kbase + nt * 16 + ln;
                    if (col > row) x = -1e30f;
                }
                s[nt][r] = x;
            }
        }

        // online softmax, per row (rows r / r+8 live in lane halves)
#pragma unroll
        for (int r = 0; r < 8; ++r) {
            float mx = fmaxf(fmaxf(s[0][r], s[1][r]), fmaxf(s[2][r], s[3][r]));
#pragma unroll
            for (int m = 8; m >= 1; m >>= 1) mx = fmaxf(mx, __shfl_xor(mx, m, 32));
            const float mnew = fmaxf(mi[r], mx);
            const float corr = __expf(mi[r] - mnew);
            float rs = 0.0f;
#pragma unroll
            for (int nt = 0; nt < 4; ++nt) {
                const float p = __expf(s[nt][r] - mnew);
                s[nt][r] = p;
                rs += p;
            }
#pragma unroll
            for (int m = 8; m >= 1; m >>= 1) rs += __shfl_xor(rs, m, 32);
            li[r] = li[r] * corr + rs;
            mi[r] = mnew;
#pragma unroll
            for (int dt = 0; dt < 8; ++dt) o[dt][r] *= corr;
        }

        // P: C-layout f32 -> bf16 A-layout via per-wave LDS scratch
#pragma unroll
        for (int nt = 0; nt < 4; ++nt)
#pragma unroll
            for (int r = 0; r < 8; ++r)
                Pt[wave][r + h * 8][nt * 16 + ln] = to_bf16(s[nt][r]);
        // same-wave DS ordering guarantees store->load visibility (DScnt)

        Frag pf[2];
#pragma unroll
        for (int c = 0; c < 2; ++c) {
            const unsigned short* pr = &Pt[wave][ln][c * 32];
            pf[c].q[0] = *(const u32x4*)(pr + h * 8);
            pf[c].q[1] = *(const u32x4*)(pr + 16 + h * 8);
        }

        // O(16x128) += P(16x64) * V(64x128)   (groups of 4 independent WMMAs)
#pragma unroll
        for (int c = 0; c < 2; ++c) {
#pragma unroll
            for (int g = 0; g < 2; ++g) {
                Frag vf[4];
#pragma unroll
                for (int i = 0; i < 4; ++i) {
                    const int dt = g * 4 + i;
                    const unsigned short* vr = &Vt[dt * 16 + ln][c * 32 + h * 8];
                    vf[i].q[0] = *(const u32x4*)(vr);
                    vf[i].q[1] = *(const u32x4*)(vr + 16);
                }
#pragma unroll
                for (int i = 0; i < 4; ++i)
                    o[g * 4 + i] = wmma_bf16(pf[c].v, vf[i].v, o[g * 4 + i]);
            }
        }
    }

    // normalize + store ctx (bf16, [B,S,H*D] layout)
#pragma unroll
    for (int dt = 0; dt < 8; ++dt) {
#pragma unroll
        for (int r = 0; r < 8; ++r) {
            const int row = b * S + qbase + wave * 16 + r + h * 8;
            const int col = head * D + dt * 16 + ln;
            Ctx[(size_t)row * HID + col] = to_bf16(o[dt][r] / li[r]);
        }
    }
}

// ---------------------------------------------------------------------------
// Launch
// ---------------------------------------------------------------------------
extern "C" void kernel_launch(void* const* d_in, const int* in_sizes, int n_in,
                              void* d_out, int out_size, void* d_ws, size_t ws_size,
                              hipStream_t stream) {
    const int B = 2, S = 2048, HID = 2048, D = 128, H = 16;
    const int M = B * S;  // 4096

    const float* x  = (const float*)d_in[0];
    // d_in[1] = mask (causal, applied analytically)
    const float* Wq = (const float*)d_in[2];
    const float* bq = (const float*)d_in[3];
    const float* Wk = (const float*)d_in[4];
    const float* bk = (const float*)d_in[5];
    const float* Wv = (const float*)d_in[6];
    const float* bv = (const float*)d_in[7];
    const float* Wo = (const float*)d_in[8];
    const float* bo = (const float*)d_in[9];
    float* out = (float*)d_out;

    // workspace layout (bf16 raw-bit arrays), 256B aligned
    size_t off = 0;
    auto alloc = [&](size_t elems) {
        void* p = (char*)d_ws + off;
        off += (elems * 2 + 255) & ~(size_t)255;
        return (unsigned short*)p;
    };
    unsigned short* xb  = alloc((size_t)M * HID);
    unsigned short* Wqb = alloc((size_t)HID * HID);
    unsigned short* Wkb = alloc((size_t)D * HID);
    unsigned short* Wvb = alloc((size_t)D * HID);
    unsigned short* Wob = alloc((size_t)HID * HID);
    unsigned short* qb  = alloc((size_t)M * HID);
    unsigned short* kb  = alloc((size_t)M * D);
    unsigned short* vb  = alloc((size_t)M * D);
    unsigned short* cb  = alloc((size_t)M * HID);
    if (off > ws_size) return;  // insufficient scratch

    auto cvt = [&](const float* s, unsigned short* d, int n) {
        int blocks = (n / 4 + 255) / 256;
        cvt_bf16_kernel<<<blocks, 256, 0, stream>>>(s, d, n);
    };
    cvt(x,  xb,  M * HID);
    cvt(Wq, Wqb, HID * HID);
    cvt(Wk, Wkb, D * HID);
    cvt(Wv, Wvb, D * HID);
    cvt(Wo, Wob, HID * HID);

    // Q = x @ Wq^T + bq  -> bf16 [M, HID]
    gemm_bf16_kernel<<<dim3(M / 128, HID / 64), 256, 0, stream>>>(
        xb, Wqb, bq, qb, nullptr, M, HID, HID);
    // K = x @ Wk^T + bk  -> bf16 [M, D]
    gemm_bf16_kernel<<<dim3(M / 128, D / 64), 256, 0, stream>>>(
        xb, Wkb, bk, kb, nullptr, M, D, HID);
    // V = x @ Wv^T + bv  -> bf16 [M, D]
    gemm_bf16_kernel<<<dim3(M / 128, D / 64), 256, 0, stream>>>(
        xb, Wvb, bv, vb, nullptr, M, D, HID);

    // attention -> ctx bf16 [M, HID]
    mqa_attn_kernel<<<dim3(S / 64, H, B), 128, 0, stream>>>(qb, kb, vb, cb);

    // out = ctx @ Wo^T + bo -> f32 [M, HID]
    gemm_bf16_kernel<<<dim3(M / 128, HID / 64), 256, 0, stream>>>(
        cb, Wob, bo, nullptr, out, M, HID, HID);
}
